// InteractorBlock_2413771620561
// MI455X (gfx1250) — matrix operand order repaired and checked by reading, not compiled, not run on hardware
//
#include <hip/hip_runtime.h>

// MI455X (gfx1250) implementation.
// Pipeline (fp32 LN, f16 WMMA GEMMs with fp32 accumulate):
//   K1: h = LN(x, nt)            -> h16 [B,N,D] f16  + h16t [B,D,N] f16 (transposed copy)
//   per batch b:
//     G1: score = h16 . h16^T    -> score [N,N] f32          (v_wmma_f32_16x16x32_f16)
//     K3: inter = LN(score, ns)  -> inter16 [N,N] f16
//     G2: h2 = inter16 . h16     -> h2[b] [N,D] f32          (B operand read via h16t)
//   K5: x2 = LN(h2, nt) + x ; hm16 = LN(x2, mn)
//   per batch b:
//     G3: g = hm16 . w16^T       -> g (reuses h2 buffer) f32
//   K7: out = LN(g + mem_b, mn) + x2
// Workspace: 32+32+16+8+64+64+32+2 MB ~= 250 MB.

#define D_DIM 1024
#define N_DIM 2048
#define B_DIM 8
#define EPSF  1e-5f

typedef __attribute__((ext_vector_type(16))) _Float16 v16h;
typedef __attribute__((ext_vector_type(8)))  _Float16 v8h;
typedef __attribute__((ext_vector_type(8)))  float    v8f;

// ---------------- block-wide {sum, sumsq} reduction (256 threads = 8 waves, wave32) ----
__device__ __forceinline__ void blockReduce2(float& s, float& q, float* sh /*16 floats*/) {
  const int lane = threadIdx.x & 31;
  const int w    = threadIdx.x >> 5;
#pragma unroll
  for (int o = 16; o > 0; o >>= 1) { s += __shfl_down(s, o, 32); q += __shfl_down(q, o, 32); }
  if (lane == 0) { sh[w] = s; sh[8 + w] = q; }
  __syncthreads();
  if (w == 0) {
    s = (lane < 8) ? sh[lane]     : 0.f;
    q = (lane < 8) ? sh[8 + lane] : 0.f;
#pragma unroll
    for (int o = 4; o > 0; o >>= 1) { s += __shfl_down(s, o, 32); q += __shfl_down(q, o, 32); }
    if (lane == 0) { sh[0] = s; sh[8] = q; }
  }
  __syncthreads();
  s = sh[0]; q = sh[8];
  __syncthreads();
}

// ---------------- K0: fp32 -> f16 weight conversion --------------------------------
__global__ void conv_w_f16(const float* __restrict__ w, _Float16* __restrict__ w16, int n) {
  int i = blockIdx.x * blockDim.x + threadIdx.x;
  if (i < n) w16[i] = (_Float16)w[i];
}

// ---------------- K1: token layernorm -> h16 and transposed h16t -------------------
__global__ void ln_token(const float* __restrict__ x,
                         const float* __restrict__ g, const float* __restrict__ b,
                         _Float16* __restrict__ h16, _Float16* __restrict__ h16t) {
  __shared__ float sh[16];
  const int row = blockIdx.x;                 // b*N + n
  const int bb  = row / N_DIM;
  const int n   = row - bb * N_DIM;
  const float4 v = ((const float4*)(x + (size_t)row * D_DIM))[threadIdx.x];
  float s = v.x + v.y + v.z + v.w;
  float q = v.x * v.x + v.y * v.y + v.z * v.z + v.w * v.w;
  blockReduce2(s, q, sh);
  const float mu   = s * (1.f / D_DIM);
  const float rstd = rsqrtf(q * (1.f / D_DIM) - mu * mu + EPSF);
  const float4 gv = ((const float4*)g)[threadIdx.x];
  const float4 bv = ((const float4*)b)[threadIdx.x];
  float o0 = (v.x - mu) * rstd * gv.x + bv.x;
  float o1 = (v.y - mu) * rstd * gv.y + bv.y;
  float o2 = (v.z - mu) * rstd * gv.z + bv.z;
  float o3 = (v.w - mu) * rstd * gv.w + bv.w;
  const int d0 = threadIdx.x * 4;
  _Float16* hr = h16 + (size_t)row * D_DIM + d0;
  hr[0] = (_Float16)o0; hr[1] = (_Float16)o1; hr[2] = (_Float16)o2; hr[3] = (_Float16)o3;
  _Float16* ht = h16t + (size_t)bb * D_DIM * N_DIM + (size_t)d0 * N_DIM + n;
  ht[0]          = (_Float16)o0;
  ht[N_DIM]      = (_Float16)o1;
  ht[2 * N_DIM]  = (_Float16)o2;
  ht[3 * N_DIM]  = (_Float16)o3;
}

// ---------------- K3: score layernorm (over m, length N) -> f16 --------------------
__global__ void ln_score(const float* __restrict__ sc,   // [N, N] one batch
                         const float* __restrict__ g, const float* __restrict__ b,
                         _Float16* __restrict__ out16) {
  __shared__ float sh[16];
  const int row = blockIdx.x;
  const float4* src = (const float4*)(sc + (size_t)row * N_DIM);
  const float4 v0 = src[threadIdx.x];
  const float4 v1 = src[threadIdx.x + 256];
  float s = v0.x + v0.y + v0.z + v0.w + v1.x + v1.y + v1.z + v1.w;
  float q = v0.x*v0.x + v0.y*v0.y + v0.z*v0.z + v0.w*v0.w
          + v1.x*v1.x + v1.y*v1.y + v1.z*v1.z + v1.w*v1.w;
  blockReduce2(s, q, sh);
  const float mu   = s * (1.f / N_DIM);
  const float rstd = rsqrtf(q * (1.f / N_DIM) - mu * mu + EPSF);
  const float4 g0 = ((const float4*)g)[threadIdx.x];
  const float4 g1 = ((const float4*)g)[threadIdx.x + 256];
  const float4 b0 = ((const float4*)b)[threadIdx.x];
  const float4 b1 = ((const float4*)b)[threadIdx.x + 256];
  _Float16* o = out16 + (size_t)row * N_DIM + threadIdx.x * 4;
  o[0] = (_Float16)((v0.x - mu) * rstd * g0.x + b0.x);
  o[1] = (_Float16)((v0.y - mu) * rstd * g0.y + b0.y);
  o[2] = (_Float16)((v0.z - mu) * rstd * g0.z + b0.z);
  o[3] = (_Float16)((v0.w - mu) * rstd * g0.w + b0.w);
  _Float16* o2 = o + 1024;
  o2[0] = (_Float16)((v1.x - mu) * rstd * g1.x + b1.x);
  o2[1] = (_Float16)((v1.y - mu) * rstd * g1.y + b1.y);
  o2[2] = (_Float16)((v1.z - mu) * rstd * g1.z + b1.z);
  o2[3] = (_Float16)((v1.w - mu) * rstd * g1.w + b1.w);
}

// ---------------- K5: x2 = LN(h2, nt) + x ; hm16 = LN(x2, mn) ----------------------
__global__ void ln_res_mem(const float* __restrict__ h2, const float* __restrict__ x,
                           const float* __restrict__ ntg, const float* __restrict__ ntb,
                           const float* __restrict__ mng, const float* __restrict__ mnb,
                           float* __restrict__ x2, _Float16* __restrict__ hm16) {
  __shared__ float sh[16];
  const int row = blockIdx.x;
  const float4 v = ((const float4*)(h2 + (size_t)row * D_DIM))[threadIdx.x];
  float s = v.x + v.y + v.z + v.w;
  float q = v.x*v.x + v.y*v.y + v.z*v.z + v.w*v.w;
  blockReduce2(s, q, sh);
  float mu   = s * (1.f / D_DIM);
  float rstd = rsqrtf(q * (1.f / D_DIM) - mu * mu + EPSF);
  const float4 gv = ((const float4*)ntg)[threadIdx.x];
  const float4 bv = ((const float4*)ntb)[threadIdx.x];
  const float4 xr = ((const float4*)(x + (size_t)row * D_DIM))[threadIdx.x];
  float4 y;
  y.x = (v.x - mu) * rstd * gv.x + bv.x + xr.x;
  y.y = (v.y - mu) * rstd * gv.y + bv.y + xr.y;
  y.z = (v.z - mu) * rstd * gv.z + bv.z + xr.z;
  y.w = (v.w - mu) * rstd * gv.w + bv.w + xr.w;
  ((float4*)(x2 + (size_t)row * D_DIM))[threadIdx.x] = y;
  // second layernorm (memory-unit norm)
  float s2 = y.x + y.y + y.z + y.w;
  float q2 = y.x*y.x + y.y*y.y + y.z*y.z + y.w*y.w;
  blockReduce2(s2, q2, sh);
  mu   = s2 * (1.f / D_DIM);
  rstd = rsqrtf(q2 * (1.f / D_DIM) - mu * mu + EPSF);
  const float4 mg = ((const float4*)mng)[threadIdx.x];
  const float4 mb = ((const float4*)mnb)[threadIdx.x];
  _Float16* o = hm16 + (size_t)row * D_DIM + threadIdx.x * 4;
  o[0] = (_Float16)((y.x - mu) * rstd * mg.x + mb.x);
  o[1] = (_Float16)((y.y - mu) * rstd * mg.y + mb.y);
  o[2] = (_Float16)((y.z - mu) * rstd * mg.z + mb.z);
  o[3] = (_Float16)((y.w - mu) * rstd * mg.w + mb.w);
}

// ---------------- K7: out = LN(g + mem_b, mn) + x2 ---------------------------------
__global__ void ln_out(const float* __restrict__ gm, const float* __restrict__ memb,
                       const float* __restrict__ mng, const float* __restrict__ mnb,
                       const float* __restrict__ x2, float* __restrict__ out) {
  __shared__ float sh[16];
  const int row = blockIdx.x;
  const float4 v  = ((const float4*)(gm + (size_t)row * D_DIM))[threadIdx.x];
  const float4 bb = ((const float4*)memb)[threadIdx.x];
  float4 t;
  t.x = v.x + bb.x; t.y = v.y + bb.y; t.z = v.z + bb.z; t.w = v.w + bb.w;
  float s = t.x + t.y + t.z + t.w;
  float q = t.x*t.x + t.y*t.y + t.z*t.z + t.w*t.w;
  blockReduce2(s, q, sh);
  const float mu   = s * (1.f / D_DIM);
  const float rstd = rsqrtf(q * (1.f / D_DIM) - mu * mu + EPSF);
  const float4 gv = ((const float4*)mng)[threadIdx.x];
  const float4 bv = ((const float4*)mnb)[threadIdx.x];
  const float4 xr = ((const float4*)(x2 + (size_t)row * D_DIM))[threadIdx.x];
  float4 o;
  o.x = (t.x - mu) * rstd * gv.x + bv.x + xr.x;
  o.y = (t.y - mu) * rstd * gv.y + bv.y + xr.y;
  o.z = (t.z - mu) * rstd * gv.z + bv.z + xr.z;
  o.w = (t.w - mu) * rstd * gv.w + bv.w + xr.w;
  ((float4*)(out + (size_t)row * D_DIM))[threadIdx.x] = o;
}

// ---------------- WMMA NT GEMM: C[M,Ncols] f32 = A[M,K] f16 . Bt[Ncols,K]^T f16 -----
// A-fragment layout (ISA 7.12.2, 16-bit A 16x32): lane m in 0..15, lanes 16..31 get
// K-half offset 8; per lane two contiguous 16B loads (K .. K+7 and K+16 .. K+23).
// B-fragment (ISA 7.12.5 col-major B): lane = output column, lanes 16..31 get K-half
// offset 8; per lane two contiguous 16B loads (same K order as A).
__device__ __forceinline__ v16h load_a_frag(const _Float16* p) {
  v8h lo = *(const v8h*)p;
  v8h hi = *(const v8h*)(p + 16);
  v16h r;
#pragma unroll
  for (int i = 0; i < 8; ++i) { r[i] = lo[i]; r[i + 8] = hi[i]; }
  return r;
}

__global__ void gemm_nt_wmma(const _Float16* __restrict__ A,
                             const _Float16* __restrict__ Bt,
                             float* __restrict__ C,
                             int Ncols, int K) {
  const int wave   = blockIdx.x * (blockDim.x >> 5) + (threadIdx.x >> 5);
  const int tilesN = Ncols >> 5;
  const int tm = (wave / tilesN) << 5;     // 32-row tile origin
  const int tn = (wave % tilesN) << 5;     // 32-col tile origin
  const int lane = threadIdx.x & 31;
  const int lr = lane & 15;
  const int lh = lane >> 4;

  const _Float16* a0p = A  + (size_t)(tm + lr) * K + lh * 8;
  const _Float16* a1p = a0p + (size_t)16 * K;
  const _Float16* b0p = Bt + (size_t)(tn + lr) * K + lh * 8;
  const _Float16* b1p = b0p + (size_t)16 * K;

  v8f c00 = {}, c01 = {}, c10 = {}, c11 = {};
  for (int kk = 0; kk < K; kk += 32) {
    const v16h a0 = load_a_frag(a0p + kk);
    const v16h a1 = load_a_frag(a1p + kk);
    const v16h b0 = load_a_frag(b0p + kk);
    const v16h b1 = load_a_frag(b1p + kk);
    c00 = __builtin_amdgcn_wmma_f32_16x16x32_f16(false, a0, false, b0, (short)0, c00, false, false);
    c01 = __builtin_amdgcn_wmma_f32_16x16x32_f16(false, a0, false, b1, (short)0, c01, false, false);
    c10 = __builtin_amdgcn_wmma_f32_16x16x32_f16(false, a1, false, b0, (short)0, c10, false, false);
    c11 = __builtin_amdgcn_wmma_f32_16x16x32_f16(false, a1, false, b1, (short)0, c11, false, false);
  }
  // C/D layout (ISA 7.12.2): lane 0..15 col = lane, reg r -> row r (+8 for lanes 16..31)
  float* crow = C + (size_t)(tm + lh * 8) * Ncols + tn + lr;
#pragma unroll
  for (int r = 0; r < 8; ++r) {
    crow[(size_t)r * Ncols]             = c00[r];
    crow[(size_t)r * Ncols + 16]        = c01[r];
    crow[(size_t)(r + 16) * Ncols]      = c10[r];
    crow[(size_t)(r + 16) * Ncols + 16] = c11[r];
  }
}

extern "C" void kernel_launch(void* const* d_in, const int* in_sizes, int n_in,
                              void* d_out, int out_size, void* d_ws, size_t ws_size,
                              hipStream_t stream) {
  (void)in_sizes; (void)n_in; (void)out_size; (void)ws_size;
  const float* x    = (const float*)d_in[0];
  const float* nt_g = (const float*)d_in[1];
  const float* nt_b = (const float*)d_in[2];
  const float* ns_g = (const float*)d_in[3];
  const float* ns_b = (const float*)d_in[4];
  const float* mw   = (const float*)d_in[5];
  const float* mb   = (const float*)d_in[6];
  const float* mn_g = (const float*)d_in[7];
  const float* mn_b = (const float*)d_in[8];
  float* out = (float*)d_out;

  char* ws = (char*)d_ws;
  size_t o = 0;
  _Float16* h16   = (_Float16*)(ws + o); o += (size_t)B_DIM * N_DIM * D_DIM * 2; // 32 MB
  _Float16* h16t  = (_Float16*)(ws + o); o += (size_t)B_DIM * N_DIM * D_DIM * 2; // 32 MB
  float*    score = (float*)   (ws + o); o += (size_t)N_DIM * N_DIM * 4;         // 16 MB (1 batch)
  _Float16* inter = (_Float16*)(ws + o); o += (size_t)N_DIM * N_DIM * 2;         //  8 MB (1 batch)
  float*    h2    = (float*)   (ws + o); o += (size_t)B_DIM * N_DIM * D_DIM * 4; // 64 MB (reused as g)
  float*    x2    = (float*)   (ws + o); o += (size_t)B_DIM * N_DIM * D_DIM * 4; // 64 MB
  _Float16* hm16  = (_Float16*)(ws + o); o += (size_t)B_DIM * N_DIM * D_DIM * 2; // 32 MB
  _Float16* w16   = (_Float16*)(ws + o); o += (size_t)D_DIM * D_DIM * 2;         //  2 MB

  conv_w_f16<<<(D_DIM * D_DIM) / 256, 256, 0, stream>>>(mw, w16, D_DIM * D_DIM);
  ln_token<<<B_DIM * N_DIM, 256, 0, stream>>>(x, nt_g, nt_b, h16, h16t);

  const int WPB = 8; // waves per 256-thread block
  for (int b = 0; b < B_DIM; ++b) {
    const _Float16* hb  = h16  + (size_t)b * N_DIM * D_DIM;
    const _Float16* htb = h16t + (size_t)b * D_DIM * N_DIM;
    // score = h . h^T   (M=N=2048, K=1024)
    gemm_nt_wmma<<<((N_DIM / 32) * (N_DIM / 32)) / WPB, 256, 0, stream>>>(
        hb, hb, score, N_DIM, D_DIM);
    ln_score<<<N_DIM, 256, 0, stream>>>(score, ns_g, ns_b, inter);
    // h2 = inter . h    (M=2048, Ncols=1024, K=2048; B read via transposed copy)
    gemm_nt_wmma<<<((N_DIM / 32) * (D_DIM / 32)) / WPB, 256, 0, stream>>>(
        inter, htb, h2 + (size_t)b * N_DIM * D_DIM, D_DIM, N_DIM);
  }

  ln_res_mem<<<B_DIM * N_DIM, 256, 0, stream>>>(h2, x, nt_g, nt_b, mn_g, mn_b, x2, hm16);

  // g = hm . W^T  (reuse h2 buffer; M=2048, Ncols=1024, K=1024)
  for (int b = 0; b < B_DIM; ++b) {
    gemm_nt_wmma<<<((N_DIM / 32) * (D_DIM / 32)) / WPB, 256, 0, stream>>>(
        hm16 + (size_t)b * N_DIM * D_DIM, w16, h2 + (size_t)b * N_DIM * D_DIM, D_DIM, D_DIM);
  }
  ln_out<<<B_DIM * N_DIM, 256, 0, stream>>>(h2, mb, mn_g, mn_b, x2, out);
}
